// NonLocalBlock_6923487281629
// MI455X (gfx1250) — matrix operand order repaired and checked by reading, compiled, not run on hardware
//
#include <hip/hip_runtime.h>
#include <cstdint>
#include <cstddef>

// ---------------------------------------------------------------------------
// NonLocal block for gfx1250 (MI455X): bf16 WMMA GEMMs + flash attention.
// B=8, C=256, Ci=128, H=W=64 -> N=4096 pixels, M=1024 pooled positions.
// Attention K/V chunks staged into LDS by the Tensor Data Mover (TDM).
// ---------------------------------------------------------------------------

typedef __attribute__((ext_vector_type(16))) __bf16 v16bf;
typedef __attribute__((ext_vector_type(8)))  __bf16 v8bf;
typedef __attribute__((ext_vector_type(8)))  float  v8f;
typedef __attribute__((ext_vector_type(4)))  unsigned int u32x4;
typedef __attribute__((ext_vector_type(8)))  int i32x8;
typedef __attribute__((ext_vector_type(4)))  int i32x4;

#define B_  8
#define C_  256
#define CI_ 128
#define N_  4096
#define M_  1024

#if defined(__has_builtin)
#if __has_builtin(__builtin_amdgcn_tensor_load_to_lds)
#define USE_TDM 1
#endif
#endif

__device__ __forceinline__ v8f wmma_bf16(v16bf a, v16bf b, v8f c) {
  return __builtin_amdgcn_wmma_f32_16x16x32_bf16(false, a, false, b, (short)0, c,
                                                 false, false);
}

// A-fragment: per lane 16 contiguous halves starting at K = lh*16.
__device__ __forceinline__ v16bf ld_afrag(const __bf16* p, int lh) {
  v8bf lo = *(const v8bf*)(p + (lh << 4));
  v8bf hi = *(const v8bf*)(p + (lh << 4) + 8);
  return __builtin_shufflevector(lo, hi, 0, 1, 2, 3, 4, 5, 6, 7,
                                 8, 9, 10, 11, 12, 13, 14, 15);
}

// B-fragment: per lane 16 contiguous halves starting at K = lh*16.
__device__ __forceinline__ v16bf ld_bfrag(const __bf16* p, int lh) {
  const __bf16* q = p + (lh << 4);
  v8bf lo = *(const v8bf*)(q);
  v8bf hi = *(const v8bf*)(q + 8);
  return __builtin_shufflevector(lo, hi, 0, 1, 2, 3, 4, 5, 6, 7,
                                 8, 9, 10, 11, 12, 13, 14, 15);
}

#ifdef USE_TDM
// Raw LDS byte offset of a shared-memory pointer (generic -> AS(3) -> int).
__device__ __forceinline__ uint32_t lds_addr_of(const void* p) {
  return (uint32_t)(uintptr_t)(const __attribute__((address_space(3))) void*)p;
}

// TDM D# group 0: count=1, LDS address, 57-bit global address, type=2.
__device__ __forceinline__ u32x4 tdm_g0(uint32_t lds_addr, uint64_t gaddr) {
  u32x4 g;
  g[0] = 1u;                                            // count = 1 (valid)
  g[1] = lds_addr;                                      // lds_addr [63:32]
  g[2] = (uint32_t)(gaddr & 0xFFFFFFFFu);               // global_addr [95:64]
  g[3] = (uint32_t)((gaddr >> 32) & 0x01FFFFFFu) | (2u << 30);  // [120:96]|type
  return g;
}

// TDM D# group 1: 2-byte elements, 2D tile (tile_dim2 = 0).
__device__ __forceinline__ i32x8 tdm_g1(uint32_t tensor_d0, uint32_t tensor_d1,
                                        uint32_t tile_d0, uint32_t tile_d1,
                                        uint32_t stride0) {
  i32x8 g = {};
  g[0] = (int)(1u << 16);                              // data_size = 1 -> 2B
  g[1] = (int)((tensor_d0 & 0xFFFFu) << 16);           // tensor_dim0[15:0]
  g[2] = (int)(((tensor_d0 >> 16) & 0xFFFFu) | ((tensor_d1 & 0xFFFFu) << 16));
  g[3] = (int)(((tensor_d1 >> 16) & 0xFFFFu) | ((tile_d0 & 0xFFFFu) << 16));
  g[4] = (int)(tile_d1 & 0xFFFFu);                     // tile_dim1; tile_dim2=0
  g[5] = (int)stride0;                                 // tensor_dim0_stride lo32
  g[6] = 0;
  g[7] = 0;
  return g;
}
#endif

// ---------------------------------------------------------------------------
// Kernel 1: three 1x1-conv projections as [Ci x C] x [C x Npix] GEMMs.
// ---------------------------------------------------------------------------
__global__ void __launch_bounds__(256)
proj_kernel(const float* __restrict__ x,
            const float* __restrict__ tw, const float* __restrict__ tbias,
            const float* __restrict__ pw, const float* __restrict__ pbias,
            const float* __restrict__ gw, const float* __restrict__ gbias,
            __bf16* __restrict__ theta, __bf16* __restrict__ phif,
            __bf16* __restrict__ gfull)
{
  const int nt = blockIdx.x;
  const int b  = blockIdx.y;
  const int pr = blockIdx.z;
  const float* w  = (pr == 0) ? tw : (pr == 1) ? pw : gw;
  const float* bs = (pr == 0) ? tbias : (pr == 1) ? pbias : gbias;

  __shared__ __align__(16) __bf16 xt[64][32];   // [n][k] (transposed tile)
  const int t    = threadIdx.x;
  const int wv   = t >> 5;
  const int lane = t & 31;
  const int lh   = lane >> 4;
  const int ll   = lane & 15;
  const int ci0  = wv * 16;
  const int n0   = nt * 64;

  v8f acc[4] = {};
  for (int k0 = 0; k0 < C_; k0 += 32) {
    __syncthreads();
    #pragma unroll
    for (int i = 0; i < 8; ++i) {
      int kk = (t >> 6) + i * 4;
      int nn = t & 63;
      xt[nn][kk] = (__bf16)x[((size_t)b * C_ + (k0 + kk)) * N_ + n0 + nn];
    }
    __syncthreads();

    const float* wr = &w[(size_t)(ci0 + ll) * C_ + k0 + (lh << 4)];
    float fa[16];
    *(float4*)&fa[0]  = ((const float4*)wr)[0];
    *(float4*)&fa[4]  = ((const float4*)wr)[1];
    *(float4*)&fa[8]  = ((const float4*)wr)[2];
    *(float4*)&fa[12] = ((const float4*)wr)[3];
    v16bf af;
    #pragma unroll
    for (int e = 0; e < 16; ++e) af[e] = (__bf16)fa[e];

    v16bf bk[4];
    #pragma unroll
    for (int nb = 0; nb < 4; ++nb)
      bk[nb] = ld_bfrag(&xt[nb * 16 + ll][0], lh);
    #pragma unroll
    for (int nb = 0; nb < 4; ++nb)
      acc[nb] = wmma_bf16(af, bk[nb], acc[nb]);
  }

  float bv[8];
  *(float4*)&bv[0] = ((const float4*)&bs[ci0 + (lh << 3)])[0];
  *(float4*)&bv[4] = ((const float4*)&bs[ci0 + (lh << 3)])[1];

  if (pr == 0) {
    #pragma unroll
    for (int nb = 0; nb < 4; ++nb) {
      int n = n0 + nb * 16 + ll;
      v8bf pk;
      #pragma unroll
      for (int r = 0; r < 8; ++r) pk[r] = (__bf16)(acc[nb][r] + bv[r]);
      *(v8bf*)&theta[((size_t)b * N_ + n) * CI_ + ci0 + (lh << 3)] = pk;
    }
  } else {
    __bf16* out = (pr == 1) ? phif : gfull;
    #pragma unroll
    for (int r = 0; r < 8; ++r) {
      int cir = ci0 + (lh << 3) + r;
      #pragma unroll
      for (int nb = 0; nb < 4; ++nb) {
        int n = n0 + nb * 16 + ll;
        out[((size_t)b * CI_ + cir) * N_ + n] = (__bf16)(acc[nb][r] + bv[r]);
      }
    }
  }
}

// ---------------------------------------------------------------------------
// Kernel 2: 2x2 max pool. phi -> transposed [b][m][ci], g -> [b][ci][m].
// ---------------------------------------------------------------------------
__global__ void __launch_bounds__(256)
pool_kernel(const __bf16* __restrict__ phif, const __bf16* __restrict__ gfull,
            __bf16* __restrict__ phi, __bf16* __restrict__ gp)
{
  int idx = blockIdx.x * blockDim.x + threadIdx.x;
  if (idx >= B_ * CI_ * M_) return;
  int m  = idx & (M_ - 1);
  int ci = (idx >> 10) & (CI_ - 1);
  int b  = idx >> 17;
  int hp = m >> 5, wp = m & 31;
  size_t base = ((size_t)b * CI_ + ci) * N_ + (size_t)(hp * 2) * 64 + wp * 2;
  float p = fmaxf(fmaxf((float)phif[base], (float)phif[base + 1]),
                  fmaxf((float)phif[base + 64], (float)phif[base + 65]));
  float g = fmaxf(fmaxf((float)gfull[base], (float)gfull[base + 1]),
                  fmaxf((float)gfull[base + 64], (float)gfull[base + 65]));
  phi[((size_t)b * M_ + m) * CI_ + ci] = (__bf16)p;  // [m][ci]
  gp[((size_t)b * CI_ + ci) * M_ + m] = (__bf16)g;   // [ci][m]
}

// ---------------------------------------------------------------------------
// Kernel 3: flash attention. grid = (N/64, B), block = 128 (4 waves).
// K/V chunks staged to LDS by the TDM (wave 0 issues, TENSORcnt + barrier).
// ---------------------------------------------------------------------------
__global__ void __launch_bounds__(128)
attn_kernel(const __bf16* __restrict__ theta, const __bf16* __restrict__ phi,
            const __bf16* __restrict__ gp, __bf16* __restrict__ yb)
{
  const int b  = blockIdx.y;
  const int q0 = blockIdx.x * 64;
  const int t    = threadIdx.x;
  const int wv   = t >> 5;
  const int lane = t & 31;
  const int lh   = lane >> 4;
  const int ll   = lane & 15;
  const int qw   = q0 + wv * 16;

  __shared__ __align__(16) __bf16 ph[64][CI_];    // keys: [key][ci]
  __shared__ __align__(16) __bf16 gv[CI_][64];    // values: [ci][key]
  __shared__ __align__(16) __bf16 Pl[4][16][64];  // per-wave P staging

  const __bf16* qrow = &theta[((size_t)b * N_ + qw + ll) * CI_];
  v16bf qf[4];
  #pragma unroll
  for (int ks = 0; ks < 4; ++ks) qf[ks] = ld_afrag(qrow + ks * 32, lh);

  v8f O[8] = {};
  float mrow[8], lrow[8];
  #pragma unroll
  for (int r = 0; r < 8; ++r) { mrow[r] = -1e30f; lrow[r] = 0.0f; }

#ifdef USE_TDM
  const uint32_t lds_ph = lds_addr_of(&ph[0][0]);
  const uint32_t lds_gv = lds_addr_of(&gv[0][0]);
#endif

  for (int kc = 0; kc < 16; ++kc) {
    __syncthreads();
#ifdef USE_TDM
    if (wv == 0) {
      // phi chunk: 64 rows x 128 ci, contiguous rows (stride 128 elems)
      uint64_t phg = (uint64_t)(uintptr_t)(phi + ((size_t)b * M_ + kc * 64) * CI_);
      // g chunk: 128 ci rows x 64 keys, row stride M_ elems
      uint64_t gvg = (uint64_t)(uintptr_t)(gp + ((size_t)b * CI_) * M_ + kc * 64);
      i32x4 z4 = {};
      i32x8 z8 = {};
      __builtin_amdgcn_tensor_load_to_lds(
          tdm_g0(lds_ph, phg), tdm_g1(CI_, 64, CI_, 64, CI_), z4, z4, z8, 0);
      __builtin_amdgcn_tensor_load_to_lds(
          tdm_g0(lds_gv, gvg), tdm_g1(M_, CI_, 64, CI_, M_), z4, z4, z8, 0);
      __builtin_amdgcn_s_wait_tensorcnt(0);
    }
#else
    { // fallback staging: vectorized uint4 copies
      const uint4* src = (const uint4*)(phi + ((size_t)b * M_ + kc * 64) * CI_);
      uint4* dst = (uint4*)&ph[0][0];
      #pragma unroll
      for (int i = 0; i < 8; ++i) dst[t + i * 128] = src[t + i * 128];
      const uint4* srcg = (const uint4*)(gp + ((size_t)b * CI_ + t) * M_ + kc * 64);
      uint4* dstg = (uint4*)&gv[t][0];
      #pragma unroll
      for (int i = 0; i < 8; ++i) dstg[i] = srcg[i];
    }
#endif
    if (kc + 1 < 16) {  // prefetch next chunk (global_prefetch_b8)
      __builtin_prefetch((const char*)(phi + ((size_t)b * M_ + (kc + 1) * 64) * CI_) +
                             t * 128, 0, 3);
      __builtin_prefetch(gp + ((size_t)b * CI_ + t) * M_ + (kc + 1) * 64, 0, 3);
    }
    __syncthreads();

    // S = Q @ K^T
    v8f S[4] = {};
    #pragma unroll
    for (int ks = 0; ks < 4; ++ks) {
      v16bf bk[4];
      #pragma unroll
      for (int cb = 0; cb < 4; ++cb)
        bk[cb] = ld_bfrag(&ph[cb * 16 + ll][ks * 32], lh);
      #pragma unroll
      for (int cb = 0; cb < 4; ++cb)
        S[cb] = wmma_bf16(qf[ks], bk[cb], S[cb]);
    }

    // online softmax update
    float mnew[8];
    #pragma unroll
    for (int r = 0; r < 8; ++r) {
      float mx = fmaxf(fmaxf(S[0][r], S[1][r]), fmaxf(S[2][r], S[3][r]));
      #pragma unroll
      for (int off = 1; off < 16; off <<= 1)
        mx = fmaxf(mx, __shfl_xor(mx, off, 32));
      mnew[r] = fmaxf(mrow[r], mx);
      float sc = __expf(mrow[r] - mnew[r]);
      lrow[r] *= sc;
      #pragma unroll
      for (int cb = 0; cb < 8; ++cb) O[cb][r] *= sc;
    }

    float rs[8];
    #pragma unroll
    for (int r = 0; r < 8; ++r) rs[r] = 0.0f;
    #pragma unroll
    for (int cb = 0; cb < 4; ++cb)
      #pragma unroll
      for (int r = 0; r < 8; ++r) {
        float p = __expf(S[cb][r] - mnew[r]);
        rs[r] += p;
        Pl[wv][r + (lh << 3)][cb * 16 + ll] = (__bf16)p;
      }
    #pragma unroll
    for (int r = 0; r < 8; ++r) {
      float s = rs[r];
      #pragma unroll
      for (int off = 1; off < 16; off <<= 1) s += __shfl_xor(s, off, 32);
      lrow[r] += s;
      mrow[r] = mnew[r];
    }
    __syncthreads();

    // O += P @ V
    #pragma unroll
    for (int ks2 = 0; ks2 < 2; ++ks2) {
      v16bf af = ld_afrag(&Pl[wv][ll][ks2 * 32], lh);
      v16bf bk[8];
      #pragma unroll
      for (int cb = 0; cb < 8; ++cb)
        bk[cb] = ld_bfrag(&gv[cb * 16 + ll][ks2 * 32], lh);
      #pragma unroll
      for (int cb = 0; cb < 8; ++cb)
        O[cb] = wmma_bf16(af, bk[cb], O[cb]);
    }
  }

  // normalize + store y [b][q][ci]
  #pragma unroll
  for (int r = 0; r < 8; ++r) {
    float inv = 1.0f / lrow[r];
    int q = qw + r + (lh << 3);
    #pragma unroll
    for (int cb = 0; cb < 8; ++cb)
      yb[((size_t)b * N_ + q) * CI_ + cb * 16 + ll] = (__bf16)(O[cb][r] * inv);
  }
}

// ---------------------------------------------------------------------------
// Kernel 4: z = W(y) + bias + x (residual). GEMM [N x Ci] * [Ci x C].
// ---------------------------------------------------------------------------
__global__ void __launch_bounds__(256)
outconv_kernel(const __bf16* __restrict__ yb, const float* __restrict__ ww,
               const float* __restrict__ wb, const float* __restrict__ x,
               float* __restrict__ z)
{
  const int b  = blockIdx.y;
  const int n0 = blockIdx.x * 64;
  const int t    = threadIdx.x;
  const int wv   = t >> 5;
  const int lane = t & 31;
  const int lh   = lane >> 4;
  const int ll   = lane & 15;
  const int np   = n0 + (wv >> 1) * 16;
  const int ch   = (wv & 1) * 128;

  const __bf16* yrow = &yb[((size_t)b * N_ + np + ll) * CI_];
  v16bf af[4];
  #pragma unroll
  for (int ks = 0; ks < 4; ++ks) af[ks] = ld_afrag(yrow + ks * 32, lh);

  v8f acc[8] = {};
  #pragma unroll
  for (int ks = 0; ks < 4; ++ks) {
    v16bf bk[8];
    #pragma unroll
    for (int cb = 0; cb < 8; ++cb) {
      int c = ch + cb * 16 + ll;
      const float* wr = &ww[(size_t)c * CI_ + ks * 32 + (lh << 4)];
      float fb[16];
      *(float4*)&fb[0]  = ((const float4*)wr)[0];
      *(float4*)&fb[4]  = ((const float4*)wr)[1];
      *(float4*)&fb[8]  = ((const float4*)wr)[2];
      *(float4*)&fb[12] = ((const float4*)wr)[3];
      #pragma unroll
      for (int e = 0; e < 16; ++e) bk[cb][e] = (__bf16)fb[e];
    }
    #pragma unroll
    for (int cb = 0; cb < 8; ++cb)
      acc[cb] = wmma_bf16(af[ks], bk[cb], acc[cb]);
  }

  #pragma unroll
  for (int cb = 0; cb < 8; ++cb) {
    int c = ch + cb * 16 + ll;
    float bias = wb[c];
    size_t base = ((size_t)b * C_ + c) * N_ + np + (lh << 3);
    float4 x0 = ((const float4*)(x + base))[0];
    float4 x1 = ((const float4*)(x + base))[1];
    float4 o0 = {acc[cb][0] + bias + x0.x, acc[cb][1] + bias + x0.y,
                 acc[cb][2] + bias + x0.z, acc[cb][3] + bias + x0.w};
    float4 o1 = {acc[cb][4] + bias + x1.x, acc[cb][5] + bias + x1.y,
                 acc[cb][6] + bias + x1.z, acc[cb][7] + bias + x1.w};
    ((float4*)(z + base))[0] = o0;
    ((float4*)(z + base))[1] = o1;
  }
}

// ---------------------------------------------------------------------------
extern "C" void kernel_launch(void* const* d_in, const int* in_sizes, int n_in,
                              void* d_out, int out_size, void* d_ws, size_t ws_size,
                              hipStream_t stream) {
  (void)in_sizes; (void)n_in; (void)out_size; (void)ws_size;
  const float* x   = (const float*)d_in[0];
  const float* g_w = (const float*)d_in[1];
  const float* g_b = (const float*)d_in[2];
  const float* t_w = (const float*)d_in[3];
  const float* t_b = (const float*)d_in[4];
  const float* p_w = (const float*)d_in[5];
  const float* p_b = (const float*)d_in[6];
  const float* w_w = (const float*)d_in[7];
  const float* w_b = (const float*)d_in[8];
  float* z = (float*)d_out;

  char* ws = (char*)d_ws;
  size_t off = 0;
  __bf16* theta = (__bf16*)(ws + off); off += (size_t)B_ * N_ * CI_ * 2;
  __bf16* phif  = (__bf16*)(ws + off); off += (size_t)B_ * CI_ * N_ * 2;
  __bf16* gfull = (__bf16*)(ws + off); off += (size_t)B_ * CI_ * N_ * 2;
  __bf16* phi   = (__bf16*)(ws + off); off += (size_t)B_ * M_ * CI_ * 2;  // [b][m][ci]
  __bf16* gp    = (__bf16*)(ws + off); off += (size_t)B_ * CI_ * M_ * 2;  // [b][ci][m]
  __bf16* yb    = (__bf16*)(ws + off); off += (size_t)B_ * N_ * CI_ * 2;

  proj_kernel<<<dim3(N_ / 64, B_, 3), 256, 0, stream>>>(
      x, t_w, t_b, p_w, p_b, g_w, g_b, theta, phif, gfull);
  pool_kernel<<<(B_ * CI_ * M_ + 255) / 256, 256, 0, stream>>>(
      phif, gfull, phi, gp);
  attn_kernel<<<dim3(N_ / 64, B_), 128, 0, stream>>>(theta, phi, gp, yb);
  outconv_kernel<<<dim3(N_ / 64, B_), 256, 0, stream>>>(yb, w_w, w_b, x, z);
}
